// EmbedAlignModel_16475494547627
// MI455X (gfx1250) — compile-verified, matrix operand order checked by reading, not run on hardware
//
#include <hip/hip_runtime.h>
#include <hip/hip_bf16.h>

// ---------------------------------------------------------------------------
// EmbedAlign forward on MI455X (gfx1250): bf16 WMMA GEMMs with fused
// exp/row-sum epilogues, persistent-workgroup LSTM recurrence.
// ---------------------------------------------------------------------------

typedef __bf16 bf16;
typedef __attribute__((ext_vector_type(16))) __bf16 v16bf;
typedef __attribute__((ext_vector_type(8)))  __bf16 v8bf;
typedef __attribute__((ext_vector_type(8)))  float  v8f;

#define DIM   256
#define LXN   2048
#define LYN   2048
#define NNEG  8192
#define LDSW  40   // padded LDS row stride in bf16 elements (80B, 16B-aligned)

// Load a 16x32 bf16 A/B fragment from an LDS tile (row-major, stride LDSW).
// Layout per ISA 7.12.2 (16-bit A 16x32; dense B mirrors it with n<->m):
//   lane<16 : rows r0..r0+15, VGPR0-3 = K 0..7,  VGPR4-7 = K 16..23
//   lane>=16: same rows,      VGPR0-3 = K 8..15, VGPR4-7 = K 24..31
// => two contiguous 16-byte chunks per lane -> two ds_load_b128.
__device__ __forceinline__ v16bf lds_frag(const bf16* base, int row0, int lane) {
  int r  = row0 + (lane & 15);
  int kh = (lane >> 4) * 8;               // 0 or 8
  const bf16* p = base + r * LDSW + kh;
  v8bf lo = *(const v8bf*)(p);
  v8bf hi = *(const v8bf*)(p + 16);
  v16bf o;
#pragma unroll
  for (int i = 0; i < 8; ++i) { o[i] = lo[i]; o[i + 8] = hi[i]; }
  return o;
}

__device__ __forceinline__ float sigmf(float x) { return 1.0f / (1.0f + __expf(-x)); }

// ---------------------------------------------------------------------------
// Generic O[M,N] = act( X[M,K=256] @ W[N,256]^T + b1 + b2 )   (f32 in/out)
// Block = 256 threads (8 waves), tile = 128(M) x 64(N), K step 32.
// All 4 B fragments are preloaded before the 4 WMMAs so the ds_loads issue as
// a batch and s_wait_dscnt overlaps with matrix-core execution.
// ---------------------------------------------------------------------------
__global__ void __launch_bounds__(256)
k_gemm_bias(const float* __restrict__ X, const float* __restrict__ W,
            const float* __restrict__ b1, const float* __restrict__ b2,
            float* __restrict__ O, int M, int N, int relu)
{
  __shared__ __align__(16) bf16 xs[128 * LDSW];
  __shared__ __align__(16) bf16 wsm[64 * LDSW];
  const int tid  = threadIdx.x;
  const int lane = tid & 31;
  const int wv   = tid >> 5;
  const int m0   = blockIdx.y * 128;
  const int n0   = blockIdx.x * 64;

  v8f acc[4] = {};
  for (int kb = 0; kb < DIM; kb += 32) {
    for (int e = tid; e < 128 * 32; e += 256) {
      int r = e >> 5, c = e & 31;
      xs[r * LDSW + c] = (bf16)X[(size_t)(m0 + r) * DIM + kb + c];
    }
    for (int e = tid; e < 64 * 32; e += 256) {
      int r = e >> 5, c = e & 31;
      wsm[r * LDSW + c] = (bf16)W[(size_t)(n0 + r) * DIM + kb + c];
    }
    __syncthreads();
    v16bf a = lds_frag(xs, wv * 16, lane);
    v16bf b[4];
#pragma unroll
    for (int f = 0; f < 4; ++f) b[f] = lds_frag(wsm, f * 16, lane);
#pragma unroll
    for (int f = 0; f < 4; ++f)
      acc[f] = __builtin_amdgcn_wmma_f32_16x16x32_bf16(
          false, a, false, b[f], (short)0, acc[f], false, false);
    __syncthreads();
  }
  // C layout: lane l, elem j -> row = 8*(l/16)+j, col = l%16 (ISA 7.12.2)
#pragma unroll
  for (int f = 0; f < 4; ++f) {
    int col = n0 + f * 16 + (lane & 15);
    float bias = b1[col] + (b2 ? b2[col] : 0.0f);
#pragma unroll
    for (int j = 0; j < 8; ++j) {
      int row = m0 + wv * 16 + (lane >> 4) * 8 + j;
      float v = acc[f][j] + bias;
      if (relu) v = fmaxf(v, 0.0f);
      O[(size_t)row * N + col] = v;
    }
  }
}

// ---------------------------------------------------------------------------
// Scoring GEMM: S = Zb[2048,256] @ Wb[N,256]^T  (both bf16, S never stored)
// mode 0: den_out[row] += sum_col exp(S)           (negative-sample denom)
// mode 1: acc_out    += sum exp(S)/(exp(S)+den_in[row])   (alignment P-sum)
// ---------------------------------------------------------------------------
__global__ void __launch_bounds__(256)
k_score(const bf16* __restrict__ Zb, const bf16* __restrict__ Wb,
        const float* __restrict__ den_in, float* __restrict__ den_out,
        float* __restrict__ acc_out, int N, int mode)
{
  __shared__ __align__(16) bf16 xs[128 * LDSW];
  __shared__ __align__(16) bf16 wsm[64 * LDSW];
  __shared__ float rowv[128];
  __shared__ float blk;
  const int tid  = threadIdx.x;
  const int lane = tid & 31;
  const int wv   = tid >> 5;
  const int m0   = blockIdx.y * 128;
  const int n0   = blockIdx.x * 64;

  if (mode == 0) {
    if (tid < 128) rowv[tid] = 0.0f;
  } else {
    if (tid < 128) rowv[tid] = den_in[m0 + tid];
    if (tid == 0)  blk = 0.0f;
  }

  v8f acc[4] = {};
  for (int kb = 0; kb < DIM; kb += 32) {
    for (int e = tid; e < 128 * 32; e += 256) {
      int r = e >> 5, c = e & 31;
      xs[r * LDSW + c] = Zb[(size_t)(m0 + r) * DIM + kb + c];
    }
    for (int e = tid; e < 64 * 32; e += 256) {
      int r = e >> 5, c = e & 31;
      wsm[r * LDSW + c] = Wb[(size_t)(n0 + r) * DIM + kb + c];
    }
    __syncthreads();
    v16bf a = lds_frag(xs, wv * 16, lane);
    v16bf b[4];
#pragma unroll
    for (int f = 0; f < 4; ++f) b[f] = lds_frag(wsm, f * 16, lane);
#pragma unroll
    for (int f = 0; f < 4; ++f)
      acc[f] = __builtin_amdgcn_wmma_f32_16x16x32_bf16(
          false, a, false, b[f], (short)0, acc[f], false, false);
    __syncthreads();
  }

  if (mode == 0) {
#pragma unroll
    for (int f = 0; f < 4; ++f)
#pragma unroll
      for (int j = 0; j < 8; ++j) {
        int lr = wv * 16 + (lane >> 4) * 8 + j;
        atomicAdd(&rowv[lr], expf(acc[f][j]));    // ds_add_f32
      }
    __syncthreads();
    if (tid < 128) atomicAdd(&den_out[m0 + tid], rowv[tid]);
  } else {
    float s = 0.0f;
#pragma unroll
    for (int f = 0; f < 4; ++f)
#pragma unroll
      for (int j = 0; j < 8; ++j) {
        int lr = wv * 16 + (lane >> 4) * 8 + j;
        float e = expf(acc[f][j]);
        s += e / (e + rowv[lr]);
      }
    atomicAdd(&blk, s);
    __syncthreads();
    if (tid == 0) atomicAdd(acc_out, blk);
  }
}

// ---------------------------------------------------------------------------
// Sequential LSTM recurrence: both cells, single persistent workgroup.
// gx* already hold xt@Wih^T + b_ih + b_hh. Thread j owns gate j of each cell.
// ---------------------------------------------------------------------------
__global__ void __launch_bounds__(1024)
k_lstm(const float* __restrict__ gxf, const float* __restrict__ gxb,
       const float* __restrict__ Whf, const float* __restrict__ Whb,
       const float* __restrict__ h0, const float* __restrict__ c0,
       float* __restrict__ hs)
{
  __shared__ float hf[DIM], hb[DIM], cf[DIM], cb[DIM];
  __shared__ float gf[4 * DIM], gb[4 * DIM];
  const int tid = threadIdx.x;
  if (tid < DIM) {
    hf[tid] = h0[tid]; hb[tid] = h0[DIM + tid];
    cf[tid] = c0[tid]; cb[tid] = c0[DIM + tid];
  }
  __syncthreads();
  const float* wf = Whf + (size_t)tid * DIM;
  const float* wb = Whb + (size_t)tid * DIM;
  for (int t = 0; t < LXN; ++t) {
    float af = gxf[(size_t)t * 4 * DIM + tid];
    float ab = gxb[(size_t)t * 4 * DIM + tid];
    if (t + 1 < LXN) {                       // global_prefetch_b8 next step
      __builtin_prefetch(gxf + (size_t)(t + 1) * 4 * DIM + tid, 0, 1);
      __builtin_prefetch(gxb + (size_t)(t + 1) * 4 * DIM + tid, 0, 1);
    }
#pragma unroll 8
    for (int k = 0; k < DIM; ++k) {
      af = fmaf(hf[k], wf[k], af);
      ab = fmaf(hb[k], wb[k], ab);
    }
    gf[tid] = af; gb[tid] = ab;
    __syncthreads();
    if (tid < DIM) {
      // gate order i,f,g,o
      float i_ = sigmf(gf[tid]);
      float f_ = sigmf(gf[DIM + tid]);
      float g_ = tanhf(gf[2 * DIM + tid]);
      float o_ = sigmf(gf[3 * DIM + tid]);
      float c1 = f_ * cf[tid] + i_ * g_;
      cf[tid] = c1;
      float h1 = o_ * tanhf(c1);
      hf[tid] = h1;
      i_ = sigmf(gb[tid]);
      f_ = sigmf(gb[DIM + tid]);
      g_ = tanhf(gb[2 * DIM + tid]);
      o_ = sigmf(gb[3 * DIM + tid]);
      float c2 = f_ * cb[tid] + i_ * g_;
      cb[tid] = c2;
      float h2 = o_ * tanhf(c2);
      hb[tid] = h2;
      hs[(size_t)t * DIM + tid] = h1 + h2;
    }
    __syncthreads();
  }
}

// ---------------------------------------------------------------------------
// Gathers (row tables of width 256)
// ---------------------------------------------------------------------------
__global__ void k_gather_f32(const float* __restrict__ tab, const int* __restrict__ idx,
                             float* __restrict__ out, int n) {
  int e = blockIdx.x * 256 + threadIdx.x;
  if (e < n * DIM) out[e] = tab[(size_t)idx[e >> 8] * DIM + (e & (DIM - 1))];
}
__global__ void k_gather_bf16(const float* __restrict__ tab, const int* __restrict__ idx,
                              bf16* __restrict__ out, int n) {
  int e = blockIdx.x * 256 + threadIdx.x;
  if (e < n * DIM) out[e] = (bf16)tab[(size_t)idx[e >> 8] * DIM + (e & (DIM - 1))];
}

// ---------------------------------------------------------------------------
// sigma = softplus(sgp); z = mu + eps*sigma (f32 + bf16 copies); KL accum
// ---------------------------------------------------------------------------
__global__ void __launch_bounds__(256)
k_msz(const float* __restrict__ mu, const float* __restrict__ sgp,
      const float* __restrict__ eps, float* __restrict__ z,
      bf16* __restrict__ zb, float* __restrict__ kl_acc)
{
  __shared__ float s;
  if (threadIdx.x == 0) s = 0.0f;
  __syncthreads();
  int e = blockIdx.x * 256 + threadIdx.x;
  float m = mu[e], p = sgp[e];
  float sig = fmaxf(p, 0.0f) + log1pf(expf(-fabsf(p)));   // stable softplus
  float zz = m + eps[e] * sig;
  z[e]  = zz;
  zb[e] = (bf16)zz;
  float kl = 0.5f * (sig * sig + m * m - 1.0f - 2.0f * logf(sig));
  atomicAdd(&s, kl);
  __syncthreads();
  if (threadIdx.x == 0) atomicAdd(kl_acc, s);
}

// ---------------------------------------------------------------------------
// num_x term: one wave32 per token; wave-shuffle dot reduction.
// ---------------------------------------------------------------------------
__global__ void __launch_bounds__(256)
k_numx(const float* __restrict__ z, const float* __restrict__ ax,
       const int* __restrict__ xi, const float* __restrict__ den, float* __restrict__ acc)
{
  int i    = blockIdx.x * 8 + (threadIdx.x >> 5);
  int lane = threadIdx.x & 31;
  const float* zi = z + (size_t)i * DIM;
  const float* ai = ax + (size_t)xi[i] * DIM;
  float s = 0.0f;
  for (int k = lane; k < DIM; k += 32) s += zi[k] * ai[k];
#pragma unroll
  for (int off = 16; off > 0; off >>= 1) s += __shfl_down(s, off, 32);
  if (lane == 0) {
    float num = expf(s);
    atomicAdd(acc, num / (num + den[i]));
  }
}

__global__ void k_zero(float* denx, float* deny, float* acc) {
  int e = blockIdx.x * 256 + threadIdx.x;
  if (e < LXN) { denx[e] = 0.0f; deny[e] = 0.0f; }
  if (e < 4) acc[e] = 0.0f;
}

__global__ void k_final(const float* __restrict__ acc, float* __restrict__ out) {
  // acc[0] = log_xi_sum, acc[1] = sum(P), acc[2] = kl_sum
  out[0] = -(acc[0] + acc[1] / (float)LXN - acc[2]);
}

// ---------------------------------------------------------------------------
extern "C" void kernel_launch(void* const* d_in, const int* in_sizes, int n_in,
                              void* d_out, int out_size, void* d_ws, size_t ws_size,
                              hipStream_t stream)
{
  (void)in_sizes; (void)n_in; (void)out_size; (void)ws_size;

  // params dict order, then x, y, neg_x, neg_y, h0, c0, eps
  const float* emb_x    = (const float*)d_in[0];
  const float* approx_x = (const float*)d_in[1];
  const float* approx_y = (const float*)d_in[2];
  const float* W_ih_f   = (const float*)d_in[3];
  const float* W_hh_f   = (const float*)d_in[4];
  const float* b_ih_f   = (const float*)d_in[5];
  const float* b_hh_f   = (const float*)d_in[6];
  const float* W_ih_b   = (const float*)d_in[7];
  const float* W_hh_b   = (const float*)d_in[8];
  const float* b_ih_b   = (const float*)d_in[9];
  const float* b_hh_b   = (const float*)d_in[10];
  const float* mu_W1    = (const float*)d_in[11];
  const float* mu_b1    = (const float*)d_in[12];
  const float* mu_W2    = (const float*)d_in[13];
  const float* mu_b2    = (const float*)d_in[14];
  const float* sig_W1   = (const float*)d_in[15];
  const float* sig_b1   = (const float*)d_in[16];
  const float* sig_W2   = (const float*)d_in[17];
  const float* sig_b2   = (const float*)d_in[18];
  const int*   x        = (const int*)d_in[19];
  const int*   y        = (const int*)d_in[20];
  const int*   neg_x    = (const int*)d_in[21];
  const int*   neg_y    = (const int*)d_in[22];
  const float* h0       = (const float*)d_in[23];
  const float* c0       = (const float*)d_in[24];
  const float* eps      = (const float*)d_in[25];

  char* w = (char*)d_ws;
  float* XE   = (float*)(w);                         //  2 MB  emb_x[x]
  float* GXF  = (float*)(w + (2u  << 20));           //  8 MB  xe@Wih_f^T + b
  float* GXB  = (float*)(w + (10u << 20));           //  8 MB
  float* HS   = (float*)(w + (18u << 20));           //  2 MB  hf+hb
  float* H1   = (float*)(w + (20u << 20));           //  2 MB  relu hidden
  float* MU   = (float*)(w + (22u << 20));           //  2 MB
  float* H2   = (float*)(w + (24u << 20));           //  2 MB
  float* SGP  = (float*)(w + (26u << 20));           //  2 MB  pre-softplus
  float* Z    = (float*)(w + (28u << 20));           //  2 MB
  bf16*  AXN  = (bf16*)(w + (30u << 20));            //  4 MB  approx_x[neg_x]
  bf16*  AYN  = (bf16*)(w + (34u << 20));            //  4 MB  approx_y[neg_y]
  bf16*  AYP  = (bf16*)(w + (38u << 20));            //  1 MB  approx_y[y]
  bf16*  ZB   = (bf16*)(w + (39u << 20));            //  1 MB  z in bf16
  float* DENX = (float*)(w + (40u << 20));           //  8 KB
  float* DENY = (float*)(w + (40u << 20) + 8192);    //  8 KB
  float* ACC  = (float*)(w + (40u << 20) + 16384);   //  16 B

  dim3 blk(256);

  k_zero<<<8, blk, 0, stream>>>(DENX, DENY, ACC);

  // 1) embedding gather
  k_gather_f32<<<(LXN * DIM) / 256, blk, 0, stream>>>(emb_x, x, XE, LXN);

  // 2) batched input-gate GEMMs (bias = b_ih + b_hh)
  k_gemm_bias<<<dim3(1024 / 64, LXN / 128), blk, 0, stream>>>(XE, W_ih_f, b_ih_f, b_hh_f, GXF, LXN, 4 * DIM, 0);
  k_gemm_bias<<<dim3(1024 / 64, LXN / 128), blk, 0, stream>>>(XE, W_ih_b, b_ih_b, b_hh_b, GXB, LXN, 4 * DIM, 0);

  // 3) sequential recurrence
  k_lstm<<<1, 1024, 0, stream>>>(GXF, GXB, W_hh_f, W_hh_b, h0, c0, HS);

  // 4) mu / sigma MLPs
  k_gemm_bias<<<dim3(DIM / 64, LXN / 128), blk, 0, stream>>>(HS, mu_W1, mu_b1, nullptr, H1, LXN, DIM, 1);
  k_gemm_bias<<<dim3(DIM / 64, LXN / 128), blk, 0, stream>>>(H1, mu_W2, mu_b2, nullptr, MU, LXN, DIM, 0);
  k_gemm_bias<<<dim3(DIM / 64, LXN / 128), blk, 0, stream>>>(HS, sig_W1, sig_b1, nullptr, H2, LXN, DIM, 1);
  k_gemm_bias<<<dim3(DIM / 64, LXN / 128), blk, 0, stream>>>(H2, sig_W2, sig_b2, nullptr, SGP, LXN, DIM, 0);

  // 5) z = mu + eps*softplus(sgp) (f32 + bf16), KL accumulation
  k_msz<<<(LXN * DIM) / 256, blk, 0, stream>>>(MU, SGP, eps, Z, ZB, &ACC[2]);

  // 6) gather negative/positive sample embeddings as bf16
  k_gather_bf16<<<(NNEG * DIM) / 256, blk, 0, stream>>>(approx_x, neg_x, AXN, NNEG);
  k_gather_bf16<<<(NNEG * DIM) / 256, blk, 0, stream>>>(approx_y, neg_y, AYN, NNEG);
  k_gather_bf16<<<(LYN  * DIM) / 256, blk, 0, stream>>>(approx_y, y,    AYP, LYN);

  // 7) denominators: rowsum exp(Z @ A^T)
  k_score<<<dim3(NNEG / 64, LXN / 128), blk, 0, stream>>>(ZB, AXN, nullptr, DENX, nullptr, NNEG, 0);
  k_score<<<dim3(NNEG / 64, LXN / 128), blk, 0, stream>>>(ZB, AYN, nullptr, DENY, nullptr, NNEG, 0);

  // 8) num_x / (num_x + den_x) sum
  k_numx<<<LXN / 8, blk, 0, stream>>>(Z, approx_x, x, DENX, &ACC[0]);

  // 9) alignment P-sum over [2048 x 2048] scores (never materialized)
  k_score<<<dim3(LYN / 64, LXN / 128), blk, 0, stream>>>(ZB, AYP, DENY, nullptr, &ACC[1], LYN, 1);

  // 10) loss
  k_final<<<1, 1, 0, stream>>>(ACC, (float*)d_out);
}
